// SurfaceFeaturePropagationCD_14259291422780
// MI455X (gfx1250) — compile-verified, matrix operand order checked by reading, not run on hardware
//
#include <hip/hip_runtime.h>
#include <stdint.h>
#include <stddef.h>

// ---------------------------------------------------------------------------
// Types for CDNA5 WMMA (wave32): v_wmma_f32_16x16x32_bf16
// ---------------------------------------------------------------------------
typedef __attribute__((ext_vector_type(16))) __bf16 v16bf;
typedef __attribute__((ext_vector_type(8)))  __bf16 v8bf;
typedef __attribute__((ext_vector_type(4)))  __bf16 v4bf;
typedef __attribute__((ext_vector_type(8)))  float  v8f;

struct alignas(16) F4 { float v[4]; };

#define CN1   262144   // dense points
#define CN2   65536    // coarse points
#define CPREV 512
#define CSKIP 256
#define CM0   256
#define CM1   256
#define RED_BLOCKS 64  // stage-2 tree reduce width

// ---------------------------------------------------------------------------
// Fragment loaders.
// 16-bit A/B fragment layout (ISA 7.12.2): lane<16 holds K {kb..kb+7, kb+16..kb+23}
// with kb=k0; lane>=16 same with kb=k0+8.  Two contiguous 16B loads per lane.
// ---------------------------------------------------------------------------
__device__ __forceinline__ v16bf load_frag_bf16(const __bf16* __restrict__ p) {
    v8bf lo = *(const v8bf*)(p);
    v8bf hi = *(const v8bf*)(p + 16);
    return __builtin_shufflevector(lo, hi, 0,1,2,3,4,5,6,7,8,9,10,11,12,13,14,15);
}

__device__ __forceinline__ v16bf load_frag_f32(const float* __restrict__ p) {
    F4 f0 = *(const F4*)(p);
    F4 f1 = *(const F4*)(p + 4);
    F4 f2 = *(const F4*)(p + 16);
    F4 f3 = *(const F4*)(p + 20);
    v16bf r;
    #pragma unroll
    for (int i = 0; i < 4; ++i) {
        r[i]      = (__bf16)f0.v[i];
        r[i + 4]  = (__bf16)f1.v[i];
        r[i + 8]  = (__bf16)f2.v[i];
        r[i + 12] = (__bf16)f3.v[i];
    }
    return r;
}

// ---------------------------------------------------------------------------
// GEMM with fused deterministic column statistics:
//   C[M,N] = A[M,K] x B + bias  (B pre-transposed bf16 BT[N,K])
//   psum/psumsq[row_block][col] = per-128-row-block column sum / sum-of-squares
// Block = 256 thr = 8 waves; WG tile 128x128; wave tile 32x64 (2x4 WMMA tiles).
// ---------------------------------------------------------------------------
template <bool AF32>
__global__ __launch_bounds__(256)
void gemm_bf16_wmma(const void* __restrict__ Av,
                    const __bf16* __restrict__ BT,
                    const float* __restrict__ bias,
                    float* __restrict__ C,
                    float* __restrict__ psum,
                    float* __restrict__ psumsq,
                    int M, int N, int K)
{
    const int lane  = threadIdx.x & 31;
    const int wave  = threadIdx.x >> 5;
    const int wm    = wave >> 1;          // 0..3  (row wave)
    const int wn    = wave & 1;           // 0..1  (col wave)
    const int lmod  = lane & 15;
    const int lhalf = lane >> 4;
    const int mBase = blockIdx.x * 128 + wm * 32;
    const int nBase = blockIdx.y * 128 + wn * 64;
    const int r0 = mBase + lmod;
    const int r1 = r0 + 16;

    const __bf16* __restrict__ Ab = (const __bf16*)Av;
    const float*  __restrict__ Af = (const float*)Av;

    v8f zero = {0.f,0.f,0.f,0.f,0.f,0.f,0.f,0.f};
    v8f acc[2][4];
    #pragma unroll
    for (int i = 0; i < 2; ++i)
        #pragma unroll
        for (int j = 0; j < 4; ++j) acc[i][j] = zero;

    for (int k0 = 0; k0 < K; k0 += 32) {
        const int kb = k0 + lhalf * 8;

        // unconditional speculative prefetch of the A stream 2 K-blocks ahead
        if constexpr (AF32) {
            __builtin_prefetch(Af + (size_t)r0 * K + k0 + 64, 0, 0);
            __builtin_prefetch(Af + (size_t)r1 * K + k0 + 64, 0, 0);
        } else {
            __builtin_prefetch(Ab + (size_t)r0 * K + k0 + 64, 0, 0);
            __builtin_prefetch(Ab + (size_t)r1 * K + k0 + 64, 0, 0);
        }

        v16bf a0, a1;
        if constexpr (AF32) {
            a0 = load_frag_f32(Af + (size_t)r0 * K + kb);
            a1 = load_frag_f32(Af + (size_t)r1 * K + kb);
        } else {
            a0 = load_frag_bf16(Ab + (size_t)r0 * K + kb);
            a1 = load_frag_bf16(Ab + (size_t)r1 * K + kb);
        }

        // issue all 4 B-fragment load pairs first (one clause, overlapped latency)
        v16bf bfr[4];
        #pragma unroll
        for (int j = 0; j < 4; ++j) {
            bfr[j] = load_frag_bf16(BT + (size_t)(nBase + j * 16 + lmod) * K + kb);
        }

        // 8 back-to-back WMMAs on the XDL pipe
        #pragma unroll
        for (int j = 0; j < 4; ++j) {
            acc[0][j] = __builtin_amdgcn_wmma_f32_16x16x32_bf16(
                false, a0, false, bfr[j], (short)0, acc[0][j], false, false);
            acc[1][j] = __builtin_amdgcn_wmma_f32_16x16x32_bf16(
                false, a1, false, bfr[j], (short)0, acc[1][j], false, false);
        }
    }

    // ---- epilogue: bias add, store, fused column sum / sum^2 ----------------
    // C/D layout: VGPR v, lane L -> row = v + (L>>4)*8, col = L&15
    float csum[4] = {0.f, 0.f, 0.f, 0.f};
    float csq [4] = {0.f, 0.f, 0.f, 0.f};
    #pragma unroll
    for (int j = 0; j < 4; ++j) {
        const int col = nBase + j * 16 + lmod;
        const float bv = bias[col];
        #pragma unroll
        for (int i = 0; i < 2; ++i) {
            const int mrow = mBase + i * 16 + lhalf * 8;
            #pragma unroll
            for (int v = 0; v < 8; ++v) {
                const float val = acc[i][j][v] + bv;
                C[(size_t)(mrow + v) * N + col] = val;
                csum[j] += val;
                csq [j] += val * val;
            }
        }
    }

    // merge the two row-halves: lanes L and L^16 hold rows 0-7 / 8-15 of the
    // same column within this wave's 32-row band
    #pragma unroll
    for (int j = 0; j < 4; ++j) {
        csum[j] += __shfl_xor(csum[j], 16);
        csq [j] += __shfl_xor(csq [j], 16);
    }

    // cross-wave reduction over the 4 row-waves via LDS
    __shared__ float lsum[4][128];
    __shared__ float lsq [4][128];
    if (lane < 16) {
        #pragma unroll
        for (int j = 0; j < 4; ++j) {
            lsum[wm][wn * 64 + j * 16 + lmod] = csum[j];
            lsq [wm][wn * 64 + j * 16 + lmod] = csq [j];
        }
    }
    __syncthreads();
    if (threadIdx.x < 128) {
        const int t = threadIdx.x;
        const float s = lsum[0][t] + lsum[1][t] + lsum[2][t] + lsum[3][t];
        const float q = lsq [0][t] + lsq [1][t] + lsq [2][t] + lsq [3][t];
        const size_t o = (size_t)blockIdx.x * CM0 + blockIdx.y * 128 + t;
        psum  [o] = s;
        psumsq[o] = q;
    }
}

// ---------------------------------------------------------------------------
// Deterministic tree reduction of per-row-block partials:
// stage 1: 64 blocks fold MB row-blocks -> 64; stage 2: one block folds 64
// and produces the BN scale/shift.
// ---------------------------------------------------------------------------
__global__ void bn_reduce_kernel(const float* __restrict__ psum,
                                 const float* __restrict__ psumsq,
                                 int MB,
                                 float* __restrict__ psum2,
                                 float* __restrict__ psumsq2)
{
    const int c = threadIdx.x;          // 0..255
    const int b = blockIdx.x;           // 0..63
    float s = 0.f, q = 0.f;
    for (int r = b; r < MB; r += RED_BLOCKS) {
        s += psum  [(size_t)r * CM0 + c];
        q += psumsq[(size_t)r * CM0 + c];
    }
    psum2  [b * CM0 + c] = s;
    psumsq2[b * CM0 + c] = q;
}

__global__ void bn_finalize_kernel(const float* __restrict__ psum2,
                                   const float* __restrict__ psumsq2,
                                   const float* __restrict__ g,
                                   const float* __restrict__ beta,
                                   float invM,
                                   float* __restrict__ scale,
                                   float* __restrict__ shift)
{
    const int c = threadIdx.x;
    float s = 0.f, q = 0.f;
    #pragma unroll 4
    for (int b = 0; b < RED_BLOCKS; ++b) {
        s += psum2  [b * CM0 + c];
        q += psumsq2[b * CM0 + c];
    }
    const float m   = s * invM;
    const float var = q * invM - m * m;
    const float sc  = g[c] * rsqrtf(var + 1e-5f);
    scale[c] = sc;
    shift[c] = beta[c] - m * sc;
}

// ---------------------------------------------------------------------------
// Weight convert + transpose: W[K,N] f32 -> WT[N,K] bf16 (tiny; one-shot).
// ---------------------------------------------------------------------------
__global__ void wt_convert_kernel(const float* __restrict__ W, __bf16* __restrict__ WT,
                                  int K, int N)
{
    const int t = blockIdx.x * 256 + threadIdx.x;
    if (t >= K * N) return;
    const int k = t / N, n = t % N;
    WT[(size_t)n * K + k] = (__bf16)W[(size_t)k * N + n];
}

// ---------------------------------------------------------------------------
// Fused interpolation + both BN affines + skip add + ReLU -> X (bf16).
// BN of the coarse path is folded through the gather since sum(w)==1.
// Thread g handles point n = g>>6, channels [4*(g&63), +4).  P2 (67 MB) is
// L2-resident for the gather.
// ---------------------------------------------------------------------------
__global__ __launch_bounds__(256)
void fuse_interp_kernel(const float* __restrict__ P2,
                        const float* __restrict__ S,
                        const float* __restrict__ dist,
                        const int*   __restrict__ idx,
                        const float* __restrict__ scf0, const float* __restrict__ shf0,
                        const float* __restrict__ scs0, const float* __restrict__ shs0,
                        __bf16* __restrict__ X)
{
    const int g  = blockIdx.x * 256 + threadIdx.x;
    const int n  = g >> 6;
    const int c0 = (g & 63) * 4;

    const float d0 = dist[3 * n + 0];
    const float d1 = dist[3 * n + 1];
    const float d2 = dist[3 * n + 2];
    const float r0 = 1.f / (d0 + 1e-8f);
    const float r1 = 1.f / (d1 + 1e-8f);
    const float r2 = 1.f / (d2 + 1e-8f);
    const float inv = 1.f / (r0 + r1 + r2);
    const float w0 = r0 * inv, w1 = r1 * inv, w2 = r2 * inv;

    const int i0 = idx[3 * n + 0];
    const int i1 = idx[3 * n + 1];
    const int i2 = idx[3 * n + 2];

    F4 p0 = *(const F4*)(P2 + (size_t)i0 * CM0 + c0);
    F4 p1 = *(const F4*)(P2 + (size_t)i1 * CM0 + c0);
    F4 p2 = *(const F4*)(P2 + (size_t)i2 * CM0 + c0);
    F4 sv = *(const F4*)(S  + (size_t)n  * CM0 + c0);
    F4 a  = *(const F4*)(scf0 + c0);
    F4 bA = *(const F4*)(shf0 + c0);
    F4 c  = *(const F4*)(scs0 + c0);
    F4 dB = *(const F4*)(shs0 + c0);

    v4bf out;
    #pragma unroll
    for (int i = 0; i < 4; ++i) {
        const float interp = w0 * p0.v[i] + w1 * p1.v[i] + w2 * p2.v[i];
        float val = a.v[i] * interp + bA.v[i] + c.v[i] * sv.v[i] + dB.v[i];
        val = fmaxf(val, 0.f);
        out[i] = (__bf16)val;
    }
    *(v4bf*)(X + (size_t)n * CM0 + c0) = out;
}

// ---------------------------------------------------------------------------
// Final in-place BN + ReLU over d_out [N1, 256] (f32).
// ---------------------------------------------------------------------------
__global__ __launch_bounds__(256)
void final_bnrelu_kernel(float* __restrict__ Y,
                         const float* __restrict__ scale,
                         const float* __restrict__ shift)
{
    const int g  = blockIdx.x * 256 + threadIdx.x;
    const int c0 = (g & 63) * 4;
    F4* y4 = (F4*)Y;
    F4 y   = y4[g];
    F4 sc  = *(const F4*)(scale + c0);
    F4 sh  = *(const F4*)(shift + c0);
    #pragma unroll
    for (int i = 0; i < 4; ++i)
        y.v[i] = fmaxf(sc.v[i] * y.v[i] + sh.v[i], 0.f);
    y4[g] = y;
}

// ---------------------------------------------------------------------------
// Host launcher
// ---------------------------------------------------------------------------
static inline size_t ws_take(size_t& off, size_t bytes) {
    size_t cur = off;
    off = (off + bytes + 255) & ~(size_t)255;
    return cur;
}

extern "C" void kernel_launch(void* const* d_in, const int* in_sizes, int n_in,
                              void* d_out, int out_size, void* d_ws, size_t ws_size,
                              hipStream_t stream)
{
    (void)in_sizes; (void)n_in; (void)out_size; (void)ws_size;

    const float* points1 = (const float*)d_in[0];   // [N1, SKIP]
    const float* points2 = (const float*)d_in[1];   // [N2, PREV]
    const float* dist    = (const float*)d_in[2];   // [N1, 3]
    const int*   idx     = (const int*)  d_in[3];   // [N1, 3]
    const float* Wf0     = (const float*)d_in[4];   // [PREV, M0]
    const float* bf0     = (const float*)d_in[5];
    const float* gf0     = (const float*)d_in[6];
    const float* betaf0  = (const float*)d_in[7];
    const float* Ws0     = (const float*)d_in[8];   // [SKIP, M0]
    const float* bs0     = (const float*)d_in[9];
    const float* gs0     = (const float*)d_in[10];
    const float* betas0  = (const float*)d_in[11];
    const float* W1      = (const float*)d_in[12];  // [M0, M1]
    const float* b1      = (const float*)d_in[13];
    const float* g1      = (const float*)d_in[14];
    const float* beta1   = (const float*)d_in[15];

    char* ws = (char*)d_ws;
    size_t off = 0;
    __bf16* Wf0T   = (__bf16*)(ws + ws_take(off, (size_t)CM0 * CPREV * 2));
    __bf16* Ws0T   = (__bf16*)(ws + ws_take(off, (size_t)CM0 * CSKIP * 2));
    __bf16* W1T    = (__bf16*)(ws + ws_take(off, (size_t)CM1 * CM0  * 2));
    float*  P2     = (float*) (ws + ws_take(off, (size_t)CN2 * CM0 * 4));
    float*  S      = (float*) (ws + ws_take(off, (size_t)CN1 * CM0 * 4));
    __bf16* X      = (__bf16*)(ws + ws_take(off, (size_t)CN1 * CM0 * 2));
    float*  psum   = (float*) (ws + ws_take(off, (size_t)(CN1/128) * CM0 * 4));
    float*  psumsq = (float*) (ws + ws_take(off, (size_t)(CN1/128) * CM0 * 4));
    float*  psum2  = (float*) (ws + ws_take(off, (size_t)RED_BLOCKS * CM0 * 4));
    float*  psumsq2= (float*) (ws + ws_take(off, (size_t)RED_BLOCKS * CM0 * 4));
    float*  scf0   = (float*) (ws + ws_take(off, CM0 * 4));
    float*  shf0   = (float*) (ws + ws_take(off, CM0 * 4));
    float*  scs0   = (float*) (ws + ws_take(off, CM0 * 4));
    float*  shs0   = (float*) (ws + ws_take(off, CM0 * 4));
    float*  sc1    = (float*) (ws + ws_take(off, CM0 * 4));
    float*  sh1    = (float*) (ws + ws_take(off, CM0 * 4));

    float* Y = (float*)d_out;   // [N1, M1] — final GEMM writes here, finished in place

    // 1) weight transpose+convert (tiny)
    wt_convert_kernel<<<(CPREV * CM0 + 255) / 256, 256, 0, stream>>>(Wf0, Wf0T, CPREV, CM0);
    wt_convert_kernel<<<(CSKIP * CM0 + 255) / 256, 256, 0, stream>>>(Ws0, Ws0T, CSKIP, CM0);
    wt_convert_kernel<<<(CM0   * CM1 + 255) / 256, 256, 0, stream>>>(W1,  W1T,  CM0,  CM1);

    // 2) coarse path GEMM (+fused stats): P2raw = points2 @ Wf0 + bf0
    gemm_bf16_wmma<true><<<dim3(CN2 / 128, CM0 / 128), 256, 0, stream>>>(
        points2, Wf0T, bf0, P2, psum, psumsq, CN2, CM0, CPREV);
    bn_reduce_kernel<<<RED_BLOCKS, 256, 0, stream>>>(psum, psumsq, CN2 / 128, psum2, psumsq2);
    bn_finalize_kernel<<<1, 256, 0, stream>>>(psum2, psumsq2, gf0, betaf0,
                                              1.0f / (float)CN2, scf0, shf0);

    // 3) skip path GEMM (+fused stats): S = points1 @ Ws0 + bs0
    gemm_bf16_wmma<true><<<dim3(CN1 / 128, CM0 / 128), 256, 0, stream>>>(
        points1, Ws0T, bs0, S, psum, psumsq, CN1, CM0, CSKIP);
    bn_reduce_kernel<<<RED_BLOCKS, 256, 0, stream>>>(psum, psumsq, CN1 / 128, psum2, psumsq2);
    bn_finalize_kernel<<<1, 256, 0, stream>>>(psum2, psumsq2, gs0, betas0,
                                              1.0f / (float)CN1, scs0, shs0);

    // 4) fused gather/interp + both BN affines + skip + ReLU -> X (bf16)
    fuse_interp_kernel<<<(CN1 * 64) / 256, 256, 0, stream>>>(
        P2, S, dist, idx, scf0, shf0, scs0, shs0, X);

    // 5) conv layer GEMM (+fused stats) straight into d_out: Y = X @ W1 + b1
    gemm_bf16_wmma<false><<<dim3(CN1 / 128, CM1 / 128), 256, 0, stream>>>(
        X, W1T, b1, Y, psum, psumsq, CN1, CM1, CM0);
    bn_reduce_kernel<<<RED_BLOCKS, 256, 0, stream>>>(psum, psumsq, CN1 / 128, psum2, psumsq2);
    bn_finalize_kernel<<<1, 256, 0, stream>>>(psum2, psumsq2, g1, beta1,
                                              1.0f / (float)CN1, sc1, sh1);

    // 6) final BN + ReLU in place on d_out
    final_bnrelu_kernel<<<(CN1 * 64) / 256, 256, 0, stream>>>(Y, sc1, sh1);
}
